// HGNN_44418551775940
// MI455X (gfx1250) — compile-verified, matrix operand order checked by reading, not run on hardware
//
#include <hip/hip_runtime.h>
#include <hip/hip_bf16.h>
#include <cstdint>
#include <cstddef>

typedef __attribute__((ext_vector_type(16))) _Float16 v16h;
typedef __attribute__((ext_vector_type(8)))  float    v8f;

#define DEVI __device__ __forceinline__

// ---- packed weight layout (halfs) in workspace: Wt[n][k], k-minor, f16 ----
#define OFF_FA1 0        // [64][32]  (K=16 padded)
#define OFF_FA2 2048     // [64][64]
#define OFF_FB1 6144     // [64][32]  (K=8 padded)
#define OFF_FB2 8192     // [64][64]
#define OFF_FE1 12288
#define OFF_FE2 16384
#define OFF_FE3 20480
#define OFF_FV1 24576    // [64][128]
#define OFF_FV2 32768
#define OFF_FV3 36864
#define OFF_KE1 40960
#define OFF_KE2 45056
#define OFF_M11 49152
#define OFF_M12 53248
#define WPK_TOTAL 57344

// softplus, threshold 20. Raw v_exp_f32 / v_log_f32 (base-2 HW transcendentals),
// no denorm guard (1+e >= 1), branchless select keeps EXEC all-ones.
DEVI float sp(float v) {
  float e = __builtin_amdgcn_exp2f(v * 1.44269504089f);   // exp(v)
  float l = __builtin_amdgcn_logf(1.0f + e) * 0.69314718056f; // log(1+e)
  return v > 20.0f ? v : l;
}

// A/B fragment: lane holds 16 contiguous K-elements of row/col (lane&15),
// K-half selected by lane>>4  (contiguous k order, same for A and B)
DEVI v16h fragA(const _Float16* base, int stride, int lane) {
  return *(const v16h*)(base + (size_t)(lane & 15) * stride + ((lane >> 4) << 4));
}

DEVI void zacc(v8f* a) { v8f z = {}; a[0] = z; a[1] = z; a[2] = z; a[3] = z; }

// 16 rows x K  @  K x 64  -> 4 accumulator tiles
template<int K>
DEVI void gemm16x64(const _Float16* aBase, int aStride, const _Float16* Wt, int lane, v8f* acc) {
#pragma unroll
  for (int kc = 0; kc < K; kc += 32) {
    v16h a = fragA(aBase + kc, aStride, lane);
#pragma unroll
    for (int nt = 0; nt < 4; ++nt) {
      v16h b = fragA(Wt + (size_t)(nt * 16) * K + kc, K, lane);
      acc[nt] = __builtin_amdgcn_wmma_f32_16x16x32_f16(
          false, a, false, b, (short)0, acc[nt], false, false);
    }
  }
}

// D-tile element (m,n): vgpr v -> m = v + 8*(lane>>4), n = (lane&15) + 16*nt
DEVI void store_sp16(const v8f* acc, const float* __restrict__ bias,
                     _Float16* dst, int dStride, int lane) {
  const int mo = (lane >> 4) << 3, nl = lane & 15;
#pragma unroll
  for (int nt = 0; nt < 4; ++nt) {
    const float bv = bias[nt * 16 + nl];
#pragma unroll
    for (int v = 0; v < 8; ++v)
      dst[(size_t)(mo + v) * dStride + nt * 16 + nl] = (_Float16)sp(acc[nt][v] + bv);
  }
}

// ---------------- weight packing ----------------
DEVI void packOne(const float* W, _Float16* dst, int K, int N, int Kp, int idx) {
  int n = idx / Kp, k = idx - n * Kp;
  dst[idx] = (k < K) ? (_Float16)W[(size_t)k * N + n] : (_Float16)0.0f;
}

__global__ __launch_bounds__(256) void prep_kernel(
    const float* faW0, const float* faW1, const float* fbW0, const float* fbW1,
    const float* feW0, const float* feW1, const float* feW2,
    const float* fvW0, const float* fvW1, const float* fvW2,
    const float* keW0, const float* keW1, const float* m1W0, const float* m1W1,
    _Float16* wpk) {
  int i = blockIdx.x * 256 + threadIdx.x;
  if      (i < OFF_FA2)   packOne(faW0, wpk + OFF_FA1, 16, 64, 32,  i - OFF_FA1);
  else if (i < OFF_FB1)   packOne(faW1, wpk + OFF_FA2, 64, 64, 64,  i - OFF_FA2);
  else if (i < OFF_FB2)   packOne(fbW0, wpk + OFF_FB1,  8, 64, 32,  i - OFF_FB1);
  else if (i < OFF_FE1)   packOne(fbW1, wpk + OFF_FB2, 64, 64, 64,  i - OFF_FB2);
  else if (i < OFF_FE2)   packOne(feW0, wpk + OFF_FE1, 64, 64, 64,  i - OFF_FE1);
  else if (i < OFF_FE3)   packOne(feW1, wpk + OFF_FE2, 64, 64, 64,  i - OFF_FE2);
  else if (i < OFF_FV1)   packOne(feW2, wpk + OFF_FE3, 64, 64, 64,  i - OFF_FE3);
  else if (i < OFF_FV2)   packOne(fvW0, wpk + OFF_FV1, 128, 64, 128, i - OFF_FV1);
  else if (i < OFF_FV3)   packOne(fvW1, wpk + OFF_FV2, 64, 64, 64,  i - OFF_FV2);
  else if (i < OFF_KE1)   packOne(fvW2, wpk + OFF_FV3, 64, 64, 64,  i - OFF_FV3);
  else if (i < OFF_KE2)   packOne(keW0, wpk + OFF_KE1, 64, 64, 64,  i - OFF_KE1);
  else if (i < OFF_M11)   packOne(keW1, wpk + OFF_KE2, 64, 64, 64,  i - OFF_KE2);
  else if (i < OFF_M12)   packOne(m1W0, wpk + OFF_M11, 64, 64, 64,  i - OFF_M11);
  else if (i < WPK_TOTAL) packOne(m1W1, wpk + OFF_M12, 64, 64, 64,  i - OFF_M12);
}

__global__ __launch_bounds__(256) void zero_kernel(float* p, int n) {
  int i = blockIdx.x * 256 + threadIdx.x;
  if (i < n) p[i] = 0.0f;
}

__global__ __launch_bounds__(256) void apply_kernel(float* x0, float* xd, int n) {
  int i = blockIdx.x * 256 + threadIdx.x;
  if (i < n) { x0[i] += xd[i]; xd[i] = 0.0f; }
}

// ---------------- fa: [N,16] -> softplus(xW+b) -> W2+b2 -> x0 f32 [N,64] ----------------
__global__ __launch_bounds__(256) void fa_kernel(
    const float* __restrict__ xin, const _Float16* __restrict__ wpk,
    const float* __restrict__ b1, const float* __restrict__ b2,
    float* __restrict__ x0, int nRows) {
  __shared__ _Float16 bufA[128 * 32];
  __shared__ _Float16 bufT[128 * 64];
  const int tid = threadIdx.x, lane = tid & 31, m0 = (tid >> 5) * 16;
  const int r0 = blockIdx.x * 128;
  for (int i = tid; i < 128 * 32; i += 256) {
    int r = i >> 5, c = i & 31;
    int row = min(r0 + r, nRows - 1);
    bufA[i] = (c < 16) ? (_Float16)xin[(size_t)row * 16 + c] : (_Float16)0.0f;
  }
  __syncthreads();
  v8f acc[4];
  zacc(acc);
  gemm16x64<32>(&bufA[(size_t)m0 * 32], 32, wpk + OFF_FA1, lane, acc);
  store_sp16(acc, b1, &bufT[(size_t)m0 * 64], 64, lane);
  zacc(acc);
  gemm16x64<64>(&bufT[(size_t)m0 * 64], 64, wpk + OFF_FA2, lane, acc);
  const int mo = (lane >> 4) << 3, nl = lane & 15;
#pragma unroll
  for (int nt = 0; nt < 4; ++nt) {
    float bv = b2[nt * 16 + nl];
#pragma unroll
    for (int v = 0; v < 8; ++v) {
      int row = r0 + m0 + mo + v;
      if (row < nRows) x0[(size_t)row * 64 + nt * 16 + nl] = acc[nt][v] + bv;
    }
  }
}

// ---------------- fb: [E,8] -> MLP -> ea f16 [E,64] ----------------
__global__ __launch_bounds__(256) void fb_kernel(
    const float* __restrict__ ein, const _Float16* __restrict__ wpk,
    const float* __restrict__ b1, const float* __restrict__ b2,
    _Float16* __restrict__ ea, int nRows) {
  __shared__ _Float16 bufA[128 * 32];
  __shared__ _Float16 bufT[128 * 64];
  const int tid = threadIdx.x, lane = tid & 31, m0 = (tid >> 5) * 16;
  const int r0 = blockIdx.x * 128;
  for (int i = tid; i < 128 * 32; i += 256) {
    int r = i >> 5, c = i & 31;
    int row = min(r0 + r, nRows - 1);
    bufA[i] = (c < 8) ? (_Float16)ein[(size_t)row * 8 + c] : (_Float16)0.0f;
  }
  __syncthreads();
  v8f acc[4];
  zacc(acc);
  gemm16x64<32>(&bufA[(size_t)m0 * 32], 32, wpk + OFF_FB1, lane, acc);
  store_sp16(acc, b1, &bufT[(size_t)m0 * 64], 64, lane);
  zacc(acc);
  gemm16x64<64>(&bufT[(size_t)m0 * 64], 64, wpk + OFF_FB2, lane, acc);
  const int mo = (lane >> 4) << 3, nl = lane & 15;
#pragma unroll
  for (int nt = 0; nt < 4; ++nt) {
    float bv = b2[nt * 16 + nl];
#pragma unroll
    for (int v = 0; v < 8; ++v) {
      int row = r0 + m0 + mo + v;
      if (row < nRows) ea[(size_t)row * 64 + nt * 16 + nl] = (_Float16)(acc[nt][v] + bv);
    }
  }
}

// ---------------- fused message-passing step ----------------
// per edge: p = x[s]*x[r]; ea += fe(p); msg = fv([x[r], ea]); xd[r] += msg (atomic)
__global__ __launch_bounds__(256) void step_kernel(
    const float* __restrict__ x, _Float16* __restrict__ ea,
    const int* __restrict__ sIdx, const int* __restrict__ rIdx,
    const _Float16* __restrict__ wpk,
    const float* __restrict__ bfe1, const float* __restrict__ bfe2, const float* __restrict__ bfe3,
    const float* __restrict__ bfv1, const float* __restrict__ bfv2, const float* __restrict__ bfv3,
    float* __restrict__ xd, int nEdges) {
  __shared__ _Float16 bufIn[128 * 128];   // cols 0:64 = x[r], cols 64:128 = temp / ea_new
  __shared__ _Float16 bufA[128 * 64];
  __shared__ int sS[128], sR[128];
  const int tid = threadIdx.x, lane = tid & 31, m0 = (tid >> 5) * 16;
  const int e0 = blockIdx.x * 128;
  if (tid < 128) {
    int e = min(e0 + tid, nEdges - 1);
    sS[tid] = sIdx[e];
    sR[tid] = rIdx[e];
  }
  __syncthreads();
  for (int i = tid; i < 128 * 64; i += 256) {
    int el = i >> 6, c = i & 63;
    float xs = x[(size_t)sS[el] * 64 + c];
    float xr = x[(size_t)sR[el] * 64 + c];
    bufA[(size_t)el * 64 + c] = (_Float16)(xs * xr);
    bufIn[(size_t)el * 128 + c] = (_Float16)xr;
  }
  __syncthreads();

  v8f acc[4];
  const int mo = (lane >> 4) << 3, nl = lane & 15;

  // fe layer1: bufA -> bufIn[:,64:128]
  zacc(acc);
  gemm16x64<64>(&bufA[(size_t)m0 * 64], 64, wpk + OFF_FE1, lane, acc);
  store_sp16(acc, bfe1, &bufIn[(size_t)m0 * 128 + 64], 128, lane);
  // fe layer2: -> bufA
  zacc(acc);
  gemm16x64<64>(&bufIn[(size_t)m0 * 128 + 64], 128, wpk + OFF_FE2, lane, acc);
  store_sp16(acc, bfe2, &bufA[(size_t)m0 * 64], 64, lane);
  // fe layer3 + residual: -> global ea (f16) and bufIn[:,64:128]
  zacc(acc);
  gemm16x64<64>(&bufA[(size_t)m0 * 64], 64, wpk + OFF_FE3, lane, acc);
#pragma unroll
  for (int nt = 0; nt < 4; ++nt) {
    float bv = bfe3[nt * 16 + nl];
#pragma unroll
    for (int v = 0; v < 8; ++v) {
      int mm = m0 + mo + v;
      size_t gi = (size_t)(e0 + mm) * 64 + nt * 16 + nl;
      float val = acc[nt][v] + bv + (float)ea[gi];
      ea[gi] = (_Float16)val;
      bufIn[(size_t)mm * 128 + 64 + nt * 16 + nl] = (_Float16)val;
    }
  }
  // fv layer1 (K=128 concat): -> bufA
  zacc(acc);
  gemm16x64<128>(&bufIn[(size_t)m0 * 128], 128, wpk + OFF_FV1, lane, acc);
  store_sp16(acc, bfv1, &bufA[(size_t)m0 * 64], 64, lane);
  // fv layer2: -> bufIn[:,0:64]
  zacc(acc);
  gemm16x64<64>(&bufA[(size_t)m0 * 64], 64, wpk + OFF_FV2, lane, acc);
  store_sp16(acc, bfv2, &bufIn[(size_t)m0 * 128], 128, lane);
  // fv layer3: -> atomic scatter into xd
  zacc(acc);
  gemm16x64<64>(&bufIn[(size_t)m0 * 128], 128, wpk + OFF_FV3, lane, acc);
#pragma unroll
  for (int nt = 0; nt < 4; ++nt) {
    float bv = bfv3[nt * 16 + nl];
#pragma unroll
    for (int v = 0; v < 8; ++v) {
      int mm = m0 + mo + v;
      int node = sR[mm];
      unsafeAtomicAdd(&xd[(size_t)node * 64 + nt * 16 + nl], acc[nt][v] + bv);
    }
  }
}

// ---------------- head: 64->64 sp -> 64->64 sp -> dot w3 + b3 ----------------
__global__ __launch_bounds__(256) void head_kernel(
    const _Float16* __restrict__ inH, const float* __restrict__ inF,
    const _Float16* __restrict__ w12, const float* __restrict__ b1, const float* __restrict__ b2,
    const float* __restrict__ w3, const float* __restrict__ b3,
    float* __restrict__ out, int nRows) {
  __shared__ _Float16 bufA[128 * 64];
  __shared__ _Float16 bufT[128 * 64];
  __shared__ float sW3[64];
  const int tid = threadIdx.x, lane = tid & 31, m0 = (tid >> 5) * 16;
  const int r0 = blockIdx.x * 128;
  if (tid < 64) sW3[tid] = w3[tid];
  for (int i = tid; i < 128 * 64; i += 256) {
    int r = i >> 6, c = i & 63;
    int row = min(r0 + r, nRows - 1);
    bufA[i] = inH ? inH[(size_t)row * 64 + c] : (_Float16)inF[(size_t)row * 64 + c];
  }
  __syncthreads();
  v8f acc[4];
  zacc(acc);
  gemm16x64<64>(&bufA[(size_t)m0 * 64], 64, w12, lane, acc);
  store_sp16(acc, b1, &bufT[(size_t)m0 * 64], 64, lane);
  zacc(acc);
  gemm16x64<64>(&bufT[(size_t)m0 * 64], 64, w12 + 4096, lane, acc);
  store_sp16(acc, b2, &bufA[(size_t)m0 * 64], 64, lane);
  __syncthreads();
  if (tid < 128) {
    int row = r0 + tid;
    float a = b3[0];
#pragma unroll
    for (int k = 0; k < 64; ++k) a += (float)bufA[(size_t)tid * 64 + k] * sW3[k];
    if (row < nRows) out[row] = a;
  }
}

extern "C" void kernel_launch(void* const* d_in, const int* in_sizes, int n_in,
                              void* d_out, int out_size, void* d_ws, size_t ws_size,
                              hipStream_t stream) {
  (void)n_in; (void)out_size; (void)ws_size;
  const float* x_in  = (const float*)d_in[0];
  const float* eattr = (const float*)d_in[1];
  const float* nve   = (const float*)d_in[2];
  const int*   eidx  = (const int*)d_in[3];
  const float *faW0 = (const float*)d_in[4],  *fab0 = (const float*)d_in[5];
  const float *faW1 = (const float*)d_in[6],  *fab1 = (const float*)d_in[7];
  const float *fbW0 = (const float*)d_in[8],  *fbb0 = (const float*)d_in[9];
  const float *fbW1 = (const float*)d_in[10], *fbb1 = (const float*)d_in[11];
  const float *feW0 = (const float*)d_in[12], *feb0 = (const float*)d_in[13];
  const float *feW1 = (const float*)d_in[14], *feb1 = (const float*)d_in[15];
  const float *feW2 = (const float*)d_in[16], *feb2 = (const float*)d_in[17];
  const float *fvW0 = (const float*)d_in[18], *fvb0 = (const float*)d_in[19];
  const float *fvW1 = (const float*)d_in[20], *fvb1 = (const float*)d_in[21];
  const float *fvW2 = (const float*)d_in[22], *fvb2 = (const float*)d_in[23];
  const float *keW0 = (const float*)d_in[24], *keb0 = (const float*)d_in[25];
  const float *keW1 = (const float*)d_in[26], *keb1 = (const float*)d_in[27];
  const float *keW2 = (const float*)d_in[28], *keb2 = (const float*)d_in[29];
  const float *m1W0 = (const float*)d_in[30], *m1b0 = (const float*)d_in[31];
  const float *m1W1 = (const float*)d_in[32], *m1b1 = (const float*)d_in[33];
  const float *m1W2 = (const float*)d_in[34], *m1b2 = (const float*)d_in[35];

  const int N = in_sizes[0] / 16;
  const int E = in_sizes[1] / 8;
  const int* sIdx = eidx;
  const int* rIdx = eidx + E;

  char* ws = (char*)d_ws;
  _Float16* wpk = (_Float16*)ws;
  _Float16* ea  = (_Float16*)(ws + 2u * WPK_TOTAL);
  float* x0 = (float*)(ws + 2u * WPK_TOTAL + (size_t)E * 64 * 2);
  float* xd = x0 + (size_t)N * 64;
  float* out = (float*)d_out;

  const int eb = (E + 127) / 128;
  const int nb = (N + 127) / 128;
  const int nvec = N * 64;
  const int zb = (nvec + 255) / 256;

  prep_kernel<<<(WPK_TOTAL + 255) / 256, 256, 0, stream>>>(
      faW0, faW1, fbW0, fbW1, feW0, feW1, feW2, fvW0, fvW1, fvW2,
      keW0, keW1, m1W0, m1W1, wpk);
  zero_kernel<<<zb, 256, 0, stream>>>(xd, nvec);
  fa_kernel<<<nb, 256, 0, stream>>>(x_in, wpk, fab0, fab1, x0, N);
  fb_kernel<<<eb, 256, 0, stream>>>(eattr, wpk, fbb0, fbb1, ea, E);
  for (int s = 0; s < 3; ++s) {
    step_kernel<<<eb, 256, 0, stream>>>(x0, ea, sIdx, rIdx, wpk,
        feb0, feb1, feb2, fvb0, fvb1, fvb2, xd, E);
    apply_kernel<<<zb, 256, 0, stream>>>(x0, xd, nvec);
  }
  head_kernel<<<eb, 256, 0, stream>>>(ea, (const float*)nullptr, wpk + OFF_M11,
      m1b0, m1b1, m1W2, m1b2, out, E);
  head_kernel<<<nb, 256, 0, stream>>>((const _Float16*)nullptr, nve, wpk + OFF_KE1,
      keb0, keb1, keW2, keb2, out + E, N);
}
